// CRF_23673859735713
// MI455X (gfx1250) — matrix operand order repaired and checked by reading, not compiled, not run on hardware
//
#include <hip/hip_runtime.h>

// CRF NLL for MI455X (gfx1250, wave32).  B=256, S=4096, L=32.
//
// Associative-scan formulation: the forward step is the linear operator
//   alpha <- (alpha @ Texp) * diag(2^(emit/ln2)),   Texp = exp(trans).
// Phase 1 (massively parallel, 4096 waves): per (batch, 256-step chunk)
//   compute the operator product Mc = PROD_t Texp*diag(ee_t) as a chain of
//   32x32x32 GEMMs -> 4x v_wmma_f32_16x16x32_f16 per step, with a wave-scalar
//   power-of-two rescale (delayed one step, applied through the emission
//   factors -> off the critical chain; exact log2 bookkeeping).
//   Masking: lens are prefix masks -> uniform loop bound min(chunk_end, len).
// Phase 2: per batch row, fold 16 chunk matrices into alpha (VALU matvec).
// Score + final reduction: deterministic fixed-order.

#define B_  256
#define S_  4096
#define L_  32
#define CL_ 256
#define NC_ (S_ / CL_)   // 16 chunks

typedef __attribute__((ext_vector_type(16))) _Float16 v16h;
typedef __attribute__((ext_vector_type(8)))  float    v8f;

#define IVLN2 1.4426950408889634f
#define LN2_  0.6931471805599453f

static __device__ __forceinline__ float fexp2(float x) {
#if __has_builtin(__builtin_amdgcn_exp2f)
  return __builtin_amdgcn_exp2f(x);          // v_exp_f32
#else
  return exp2f(x);
#endif
}
static __device__ __forceinline__ float flog2(float x) {
#if __has_builtin(__builtin_amdgcn_logf)
  return __builtin_amdgcn_logf(x);           // v_log_f32 (base-2)
#else
  return log2f(x);
#endif
}
static __device__ __forceinline__ float max16_tree(const float* v) {
  float m0 = fmaxf(v[0], v[1]),  m1 = fmaxf(v[2], v[3]);
  float m2 = fmaxf(v[4], v[5]),  m3 = fmaxf(v[6], v[7]);
  float m4 = fmaxf(v[8], v[9]),  m5 = fmaxf(v[10], v[11]);
  float m6 = fmaxf(v[12], v[13]), m7 = fmaxf(v[14], v[15]);
  m0 = fmaxf(m0, m1); m2 = fmaxf(m2, m3); m4 = fmaxf(m4, m5); m6 = fmaxf(m6, m7);
  return fmaxf(fmaxf(m0, m2), fmaxf(m4, m6));
}

// ---------------------------------------------------------------------------
// Phase 1: chunk operator products. grid = (NC_, B_), 32 threads (one wave).
// ---------------------------------------------------------------------------
__global__ __launch_bounds__(32) void crf_chunk(
    const float* __restrict__ emit, const float* __restrict__ trans,
    const int* __restrict__ lens, float* __restrict__ mats,
    float* __restrict__ cscales)
{
  // 36-dword rows: 16B aligned; rows k and k+8 on different banks.
  alignas(16) __shared__ float qbuf[32][36];

  const int lane = threadIdx.x & 31;
  const int hi = lane >> 4, lr = lane & 15;
  const int cch = blockIdx.x;                 // chunk index
  const int b   = blockIdx.y;                 // batch row
  const int len = lens[b];
  const int t0   = cch * CL_ + (cch == 0 ? 1 : 0);
  const int te   = (cch + 1) * CL_;
  const int tend = (te < len) ? te : len;     // prefix mask -> uniform bound

  const int kb = 8 * hi;
  int Ks[16];
#pragma unroll
  for (int h = 0; h < 16; ++h) Ks[h] = kb + (h < 8 ? h : 8 + h);

  // Texp in WMMA B layout (constant), both 16-column tiles
  v16h bT0, bT1;
#pragma unroll
  for (int h = 0; h < 16; ++h) {
    bT0[h] = (_Float16)__expf(trans[Ks[h] * L_ + lr]);
    bT1[h] = (_Float16)__expf(trans[Ks[h] * L_ + lr + 16]);
  }

  // running matrix M, A layout (2 row tiles); start = identity
  v16h a0, a1;
  float v0[16], v1[16];
#pragma unroll
  for (int h = 0; h < 16; ++h) {
    v0[h] = (Ks[h] == lr)      ? 1.0f : 0.0f;
    v1[h] = (Ks[h] == 16 + lr) ? 1.0f : 0.0f;
    a0[h] = (_Float16)v0[h];
    a1[h] = (_Float16)v1[h];
  }

  const float* eb = emit + (size_t)b * S_ * L_;
  float sc = 0.0f, rdel = 1.0f, Edel = 0.0f;  // delayed scalar rescale

  for (int t = t0; t < tend; ++t) {
    // emission column factors; delayed 2^-(E+3) folded in (off the chain)
    const float ee0 = fexp2(eb[(size_t)t * L_ + lr]      * IVLN2) * rdel;
    const float ee1 = fexp2(eb[(size_t)t * L_ + lr + 16] * IVLN2) * rdel;
    __builtin_prefetch(&eb[(size_t)(t + 8) * L_ + lr], 0, 1);
    sc += Edel;                                // exact scale bookkeeping

    // M' = M @ Texp : 32x32x32 via 4 WMMAs
    v8f c00 = {}, c01 = {}, c10 = {}, c11 = {};
    c00 = __builtin_amdgcn_wmma_f32_16x16x32_f16(false, a0, false, bT0,
                                                 (short)0, c00, false, false);
    c01 = __builtin_amdgcn_wmma_f32_16x16x32_f16(false, a0, false, bT1,
                                                 (short)0, c01, false, false);
    c10 = __builtin_amdgcn_wmma_f32_16x16x32_f16(false, a1, false, bT0,
                                                 (short)0, c10, false, false);
    c11 = __builtin_amdgcn_wmma_f32_16x16x32_f16(false, a1, false, bT1,
                                                 (short)0, c11, false, false);

    // column scale by emissions, store C layout (merges to ds_store_2addr)
#pragma unroll
    for (int k = 0; k < 8; ++k) {
      float* q0 = &qbuf[k + 8 * hi][lr];
      float* q1 = &qbuf[16 + k + 8 * hi][lr];
      q0[0]  = c00[k] * ee0;
      q0[16] = c01[k] * ee1;
      q1[0]  = c10[k] * ee0;
      q1[16] = c11[k] * ee1;
    }
    __builtin_amdgcn_wave_barrier();          // same-wave DS ops are in order

    // transpose reload (A layout): 8x ds_load_b128
    const float4* bp0 = (const float4*)&qbuf[lr][0];
    const float4* bp1 = (const float4*)&qbuf[16 + lr][0];
    const float4 y0 = bp0[(kb >> 2) + 0], y1 = bp0[(kb >> 2) + 1];
    const float4 y2 = bp0[((kb + 16) >> 2) + 0], y3 = bp0[((kb + 16) >> 2) + 1];
    const float4 z0 = bp1[(kb >> 2) + 0], z1 = bp1[(kb >> 2) + 1];
    const float4 z2 = bp1[((kb + 16) >> 2) + 0], z3 = bp1[((kb + 16) >> 2) + 1];
    v0[0]=y0.x; v0[1]=y0.y; v0[2]=y0.z; v0[3]=y0.w;
    v0[4]=y1.x; v0[5]=y1.y; v0[6]=y1.z; v0[7]=y1.w;
    v0[8]=y2.x; v0[9]=y2.y; v0[10]=y2.z; v0[11]=y2.w;
    v0[12]=y3.x; v0[13]=y3.y; v0[14]=y3.z; v0[15]=y3.w;
    v1[0]=z0.x; v1[1]=z0.y; v1[2]=z0.z; v1[3]=z0.w;
    v1[4]=z1.x; v1[5]=z1.y; v1[6]=z1.z; v1[7]=z1.w;
    v1[8]=z2.x; v1[9]=z2.y; v1[10]=z2.z; v1[11]=z2.w;
    v1[12]=z3.x; v1[13]=z3.y; v1[14]=z3.z; v1[15]=z3.w;
#pragma unroll
    for (int h = 0; h < 16; ++h) {            // scale applied NEXT step via ee
      a0[h] = (_Float16)v0[h];
      a1[h] = (_Float16)v1[h];
    }

    // wave-scalar max -> delayed power-of-two rescale (off the chain)
    float mx = fmaxf(max16_tree(v0), max16_tree(v1));
#pragma unroll
    for (int off = 16; off; off >>= 1) mx = fmaxf(mx, __shfl_xor(mx, off, 32));
    const unsigned ebits = __float_as_uint(mx) >> 23;
    Edel = (float)((int)ebits - 124);                  // E+3: headroom for f16
    rdel = __uint_as_float((251u - ebits) << 23);      // 2^-(E+3)
  }

  // store normalized chunk matrix (row-major) + its log2 scale
  float* mp = mats + ((size_t)b * NC_ + cch) * (L_ * L_);
#pragma unroll
  for (int h = 0; h < 16; ++h) {
    mp[lr * L_ + Ks[h]]        = v0[h];
    mp[(16 + lr) * L_ + Ks[h]] = v1[h];
  }
  if (lane == 0) cscales[b * NC_ + cch] = sc;
}

// ---------------------------------------------------------------------------
// Phase 2: fold chunk matrices into alpha. 256 blocks x 32 threads (1 wave/b).
// ---------------------------------------------------------------------------
__global__ __launch_bounds__(32) void crf_fold(
    const float* __restrict__ emit, const float* __restrict__ strans,
    const float* __restrict__ etrans, const float* __restrict__ mats,
    const float* __restrict__ cscales, float* __restrict__ ws_logZ)
{
  const int j = threadIdx.x & 31;             // state column
  const int b = blockIdx.x;

  // alpha0 = (strans + emit[:,0,:]) / ln2 -> normalized linear + scale c
  float arg = (emit[(size_t)b * S_ * L_ + j] + strans[j]) * IVLN2;
  float mx = arg;
#pragma unroll
  for (int off = 16; off; off >>= 1) mx = fmaxf(mx, __shfl_xor(mx, off, 32));
  float q = fexp2(arg - mx);
  float c = mx;

  for (int ch = 0; ch < NC_; ++ch) {
    const float* mp = mats + ((size_t)b * NC_ + ch) * (L_ * L_) + j;
    float acc = 0.0f;
#pragma unroll
    for (int i = 0; i < 32; ++i)
      acc = fmaf(__shfl(q, i, 32), mp[i * L_], acc);   // column-j dot
    float m2 = acc;
#pragma unroll
    for (int off = 16; off; off >>= 1) m2 = fmaxf(m2, __shfl_xor(m2, off, 32));
    const unsigned eb2 = __float_as_uint(m2) >> 23;
    acc *= __uint_as_float((254u - eb2) << 23);        // 2^-E
    c += (float)((int)eb2 - 127) + cscales[b * NC_ + ch];
    q = acc;
  }

  // logZ_b = ln2 * (c + log2( sum_j q_j * 2^(etrans_j/ln2) ))
  float s = q * fexp2(etrans[j] * IVLN2);
#pragma unroll
  for (int off = 16; off; off >>= 1) s += __shfl_xor(s, off, 32);
  if (j == 0) ws_logZ[b] = LN2_ * (c + flog2(s));
}

// ---------------------------------------------------------------------------
// Gold-path score (also publishes per-row lengths for phase 1).
// ---------------------------------------------------------------------------
__global__ __launch_bounds__(256) void crf_score(
    const float* __restrict__ emit, const int* __restrict__ target,
    const unsigned char* __restrict__ mask, const float* __restrict__ trans,
    const float* __restrict__ strans, const float* __restrict__ etrans,
    float* __restrict__ ws_score, int* __restrict__ ws_len)
{
  __shared__ int   sh_i[256];
  __shared__ float sh_f[256];
  __shared__ int   s_len;

  const int b = blockIdx.x, tid = threadIdx.x;

  {
    const uint4* mp = (const uint4*)(mask + (size_t)b * S_ + tid * 16);
    uint4 w = mp[0];
    sh_i[tid] = __popc(w.x & 0x01010101u) + __popc(w.y & 0x01010101u) +
                __popc(w.z & 0x01010101u) + __popc(w.w & 0x01010101u);
  }
  __syncthreads();
  for (int off = 128; off; off >>= 1) {
    if (tid < off) sh_i[tid] += sh_i[tid + off];
    __syncthreads();
  }
  if (tid == 0) { s_len = sh_i[0]; ws_len[b] = sh_i[0]; }
  __syncthreads();
  const int len = s_len;

  float sc = 0.0f;
#pragma unroll 4
  for (int i = 0; i < 16; ++i) {
    const int t = tid + 256 * i;
    if (t < len) {
      const int tg = target[(size_t)b * S_ + t];
      sc += emit[((size_t)b * S_ + t) * L_ + tg];
      if (t > 0) sc += trans[target[(size_t)b * S_ + t - 1] * L_ + tg];
    }
  }
  sh_f[tid] = sc;
  __syncthreads();
  for (int off = 128; off; off >>= 1) {
    if (tid < off) sh_f[tid] += sh_f[tid + off];
    __syncthreads();
  }
  if (tid == 0) {
    ws_score[b] = sh_f[0] + strans[target[(size_t)b * S_]] +
                  etrans[target[(size_t)b * S_ + len - 1]];
  }
}

// ---------------------------------------------------------------------------
__global__ void crf_final(const float* __restrict__ ws_score,
                          const float* __restrict__ ws_logZ,
                          float* __restrict__ out)
{
  if (threadIdx.x == 0 && blockIdx.x == 0) {
    float lz = 0.0f, sc = 0.0f;
    for (int i = 0; i < B_; ++i) lz += ws_logZ[i];
    for (int i = 0; i < B_; ++i) sc += ws_score[i];
    out[0] = (lz - sc) / (float)B_;
  }
}

extern "C" void kernel_launch(void* const* d_in, const int* in_sizes, int n_in,
                              void* d_out, int out_size, void* d_ws, size_t ws_size,
                              hipStream_t stream) {
  const float*         emit   = (const float*)d_in[0];
  const int*           target = (const int*)d_in[1];
  const unsigned char* mask   = (const unsigned char*)d_in[2];  // jnp bool = 1B
  const float*         trans  = (const float*)d_in[3];
  const float*         strans = (const float*)d_in[4];
  const float*         etrans = (const float*)d_in[5];

  float* ws       = (float*)d_ws;
  float* ws_score = ws;                 // [0..255]
  float* ws_logZ  = ws + 256;           // [256..511]
  int*   ws_len   = (int*)(ws + 512);   // [512..767]
  float* cscales  = ws + 768;           // B*NC = 4096 floats
  float* mats     = ws + 8192;          // B*NC*1024 floats = 16 MB

  crf_score<<<B_, 256, 0, stream>>>(emit, target, mask, trans, strans, etrans,
                                    ws_score, ws_len);
  crf_chunk<<<dim3(NC_, B_), 32, 0, stream>>>(emit, trans, ws_len, mats,
                                              cscales);
  crf_fold<<<B_, 32, 0, stream>>>(emit, strans, etrans, mats, cscales,
                                  ws_logZ);
  crf_final<<<1, 1, 0, stream>>>(ws_score, ws_logZ, (float*)d_out);
}
